// AnemllQATLinear_43190191128770
// MI455X (gfx1250) — matrix operand order repaired and checked by reading, not compiled, not run on hardware
//
#include <hip/hip_runtime.h>
#include <hip/hip_bf16.h>

typedef __attribute__((ext_vector_type(16))) __bf16 v16bf;
typedef __attribute__((ext_vector_type(8)))  __bf16 v8bf;
typedef __attribute__((ext_vector_type(8)))  float  v8f;

#define IN_DIM  4096
#define OUT_DIM 4096
#define M_DIM   8192   // B*S = 4*2048
#define LORA_R  16
#define SCALING 2.0f
#define EPS_Q   1e-8f

#define BM 128
#define BN 128
#define BK 32

__device__ __forceinline__ unsigned short f32_bf16(float f) {
  unsigned u = __float_as_uint(f);
  unsigned r = u + 0x7FFFu + ((u >> 16) & 1u);   // round-to-nearest-even
  return (unsigned short)(r >> 16);
}

// ---------------------------------------------------------------------------
// Kernel 1: fake-quant weight + fold LoRA (rank-16) into it, emit bf16.
// w_eff[o,i] = lut[idx]*scales + 2.0 * sum_r lora_B[o,r]*lora_A[r,i]
// ---------------------------------------------------------------------------
__global__ __launch_bounds__(256)
void prep_weight(const float* __restrict__ weight,
                 const float* __restrict__ scale_A,   // [OUT,4]
                 const float* __restrict__ scale_B,   // [4,IN]
                 const float* __restrict__ lora_A,    // [16,IN]
                 const float* __restrict__ lora_B,    // [OUT,16]
                 const float* __restrict__ lut,       // [16]
                 unsigned short* __restrict__ wq) {   // [OUT,IN] bf16
  const int gid = blockIdx.x * blockDim.x + threadIdx.x;  // OUT*IN/8 threads
  const int o  = gid >> 9;                 // IN/8 = 512 groups per row
  const int i0 = (gid & 511) << 3;

  const float a0 = scale_A[o*4+0], a1 = scale_A[o*4+1];
  const float a2 = scale_A[o*4+2], a3 = scale_A[o*4+3];
  float lb[LORA_R];
#pragma unroll
  for (int r = 0; r < LORA_R; ++r) lb[r] = lora_B[o*LORA_R + r];

#pragma unroll
  for (int j = 0; j < 8; ++j) {
    const int i = i0 + j;
    float s = a0*scale_B[i] + a1*scale_B[IN_DIM + i]
            + a2*scale_B[2*IN_DIM + i] + a3*scale_B[3*IN_DIM + i];
    s = fmaxf(s, EPS_Q);
    const float w   = weight[(size_t)o*IN_DIM + i];
    const float nrm = fminf(fmaxf(w / s, -1.0f), 1.0f);
    int idx = (int)rintf((nrm + 1.0f) * 7.5f);      // step = 2/15
    idx = min(max(idx, 0), 15);
    const float deq = lut[idx] * s;
    float lo = 0.0f;
#pragma unroll
    for (int r = 0; r < LORA_R; ++r) lo += lb[r] * lora_A[(size_t)r*IN_DIM + i];
    wq[(size_t)o*IN_DIM + i] = f32_bf16(deq + SCALING * lo);
  }
}

// ---------------------------------------------------------------------------
// Kernel 2: x f32 -> bf16
// ---------------------------------------------------------------------------
__global__ __launch_bounds__(256)
void conv_x(const float* __restrict__ x, unsigned short* __restrict__ xb) {
  const size_t gid = ((size_t)blockIdx.x * blockDim.x + threadIdx.x) * 4;
  const float4 v = *(const float4*)(x + gid);
  xb[gid + 0] = f32_bf16(v.x);
  xb[gid + 1] = f32_bf16(v.y);
  xb[gid + 2] = f32_bf16(v.z);
  xb[gid + 3] = f32_bf16(v.w);
}

// ---------------------------------------------------------------------------
// Kernel 3: bf16 GEMM, y[m,n] = sum_k A[m,k]*W[n,k] + bias[n]
// 128x128 block, BK=32, double-buffered LDS via async copies,
// 8 waves (4 along M x 2 along N), wave tile 32x64 -> 2x4 WMMA accumulators.
// ---------------------------------------------------------------------------
__global__ __launch_bounds__(256)
void gemm_bf16(const unsigned short* __restrict__ Abf,   // [M,K] bf16
               const unsigned short* __restrict__ Wbf,   // [N,K] bf16
               const float* __restrict__ bias,
               float* __restrict__ out) {
  __shared__ unsigned short As[2][BM][BK];
  __shared__ unsigned short Bs[2][BN][BK];

  const int tid  = threadIdx.x;
  const int lane = tid & 31;
  const int wid  = tid >> 5;
  const int wm   = wid & 3;        // 0..3  (M)
  const int wn   = wid >> 2;       // 0..1  (N)
  const int mBlock = blockIdx.y * BM;
  const int nBlock = blockIdx.x * BN;

  // async copy of one K-stage: A tile 128x32 halves, B tile 128x32 halves.
  // 512 16-byte chunks per tile; each of 256 threads moves 2 chunks per tile.
  auto issueStage = [&](int buf, int k0) {
#pragma unroll
    for (int p = 0; p < 2; ++p) {
      const int chunk = tid + p * 256;       // 0..511
      const int r = chunk >> 2;
      const int c = (chunk & 3) << 3;        // half-index within row
      unsigned ldsA = (unsigned)(unsigned long long)&As[buf][r][c];
      unsigned long long gA =
          (unsigned long long)(Abf + (size_t)(mBlock + r) * IN_DIM + k0 + c);
      asm volatile("global_load_async_to_lds_b128 %0, %1, off"
                   :: "v"(ldsA), "v"(gA) : "memory");
      unsigned ldsB = (unsigned)(unsigned long long)&Bs[buf][r][c];
      unsigned long long gB =
          (unsigned long long)(Wbf + (size_t)(nBlock + r) * IN_DIM + k0 + c);
      asm volatile("global_load_async_to_lds_b128 %0, %1, off"
                   :: "v"(ldsB), "v"(gB) : "memory");
    }
  };

  v8f acc[2][4];
#pragma unroll
  for (int mt = 0; mt < 2; ++mt)
#pragma unroll
    for (int nt = 0; nt < 4; ++nt)
#pragma unroll
      for (int e = 0; e < 8; ++e) acc[mt][nt][e] = 0.0f;

  issueStage(0, 0);
  const int KTILES = IN_DIM / BK;            // 128

  for (int kt = 0; kt < KTILES; ++kt) {
    const int buf = kt & 1;
    asm volatile("s_wait_asynccnt 0x0" ::: "memory");
    __syncthreads();                          // stage kt visible to all waves
    if (kt + 1 < KTILES) issueStage(buf ^ 1, (kt + 1) * BK);

    // A fragments: 16-bit A layout -> lanes 0-15 hold K 0..7, lanes 16-31 K 8..15
    // (VGPR 0..3), then +16 halves for VGPR 4..7.  Two contiguous b128 reads
    // (compiler-issued ds_load_b128 with its own dscnt tracking).
    v16bf afrag[2];
#pragma unroll
    for (int mt = 0; mt < 2; ++mt) {
      const unsigned short* pa =
          &As[buf][wm * 32 + mt * 16 + (lane & 15)][(lane >> 4) * 8];
      v8bf alo = *(const v8bf*)pa;
      v8bf ahi = *(const v8bf*)(pa + 16);
      afrag[mt] = __builtin_shufflevector(alo, ahi,
          0,1,2,3,4,5,6,7,8,9,10,11,12,13,14,15);
    }

    // B fragments: all 8 b128 loads (4 N-tiles x 2 K-halves) issued in one
    // asm block, single s_wait_dscnt, early-clobber outputs so no address reg
    // can be clobbered by an earlier load in the block.
    const unsigned bbase = (unsigned)(unsigned long long)
        &Bs[buf][wn * 64 + (lane & 15)][(lane >> 4) * 8];
    v8bf bf0, bf1, bf2, bf3, bf4, bf5, bf6, bf7;
    asm volatile(
        "ds_load_b128 %0, %8\n\t"
        "ds_load_b128 %1, %9\n\t"
        "ds_load_b128 %2, %10\n\t"
        "ds_load_b128 %3, %11\n\t"
        "ds_load_b128 %4, %12\n\t"
        "ds_load_b128 %5, %13\n\t"
        "ds_load_b128 %6, %14\n\t"
        "ds_load_b128 %7, %15\n\t"
        "s_wait_dscnt 0x0"
        : "=&v"(bf0), "=&v"(bf1), "=&v"(bf2), "=&v"(bf3),
          "=&v"(bf4), "=&v"(bf5), "=&v"(bf6), "=&v"(bf7)
        : "v"(bbase),          "v"(bbase + 32),
          "v"(bbase + 0x400),  "v"(bbase + 0x420),
          "v"(bbase + 0x800),  "v"(bbase + 0x820),
          "v"(bbase + 0xc00),  "v"(bbase + 0xc20));

    v16bf bfrag[4];
    bfrag[0] = __builtin_shufflevector(bf0, bf1, 0,1,2,3,4,5,6,7,8,9,10,11,12,13,14,15);
    bfrag[1] = __builtin_shufflevector(bf2, bf3, 0,1,2,3,4,5,6,7,8,9,10,11,12,13,14,15);
    bfrag[2] = __builtin_shufflevector(bf4, bf5, 0,1,2,3,4,5,6,7,8,9,10,11,12,13,14,15);
    bfrag[3] = __builtin_shufflevector(bf6, bf7, 0,1,2,3,4,5,6,7,8,9,10,11,12,13,14,15);

#pragma unroll
    for (int nt = 0; nt < 4; ++nt)
#pragma unroll
      for (int mt = 0; mt < 2; ++mt)
        acc[mt][nt] = __builtin_amdgcn_wmma_f32_16x16x32_bf16(
            false, afrag[mt], false, bfrag[nt],
            (short)0, acc[mt][nt], false, false);
  }

  // Epilogue: C/D layout -> lanes 0-15: M=v, N=lane ; lanes 16-31: M=v+8, N=lane-16
#pragma unroll
  for (int mt = 0; mt < 2; ++mt) {
#pragma unroll
    for (int nt = 0; nt < 4; ++nt) {
      const int n = nBlock + wn * 64 + nt * 16 + (lane & 15);
      const float bv = bias[n];
      const int mbase = mBlock + wm * 32 + mt * 16 + ((lane >> 4) << 3);
      float* po = out + (size_t)mbase * OUT_DIM + n;
#pragma unroll
      for (int v = 0; v < 8; ++v)
        po[(size_t)v * OUT_DIM] = acc[mt][nt][v] + bv;
    }
  }
}

// ---------------------------------------------------------------------------
extern "C" void kernel_launch(void* const* d_in, const int* in_sizes, int n_in,
                              void* d_out, int out_size, void* d_ws, size_t ws_size,
                              hipStream_t stream) {
  const float* x       = (const float*)d_in[0];
  const float* weight  = (const float*)d_in[1];
  const float* scale_A = (const float*)d_in[2];
  const float* scale_B = (const float*)d_in[3];
  const float* bias    = (const float*)d_in[4];
  const float* lora_A  = (const float*)d_in[5];
  const float* lora_B  = (const float*)d_in[6];
  const float* lut     = (const float*)d_in[7];

  unsigned short* xb = (unsigned short*)d_ws;                 // [M,K] bf16 (67 MB)
  unsigned short* wq = xb + (size_t)M_DIM * IN_DIM;           // [N,K] bf16 (33.5 MB)
  float* out = (float*)d_out;

  // x -> bf16 (33.5M elems, 4 per thread)
  conv_x<<<(M_DIM * IN_DIM) / (256 * 4), 256, 0, stream>>>(x, xb);

  // fake-quant + LoRA fold -> bf16 weight
  prep_weight<<<(OUT_DIM * IN_DIM / 8) / 256, 256, 0, stream>>>(
      weight, scale_A, scale_B, lora_A, lora_B, lut, wq);

  // main GEMM
  dim3 grid(OUT_DIM / BN, M_DIM / BM);   // 32 x 64
  gemm_bf16<<<grid, 256, 0, stream>>>(xb, wq, bias, out);
}
